// EpisodicMemory_69312182223071
// MI455X (gfx1250) — matrix operand order repaired and checked by reading, not compiled, not run on hardware
//
#include <hip/hip_runtime.h>
#include <math.h>

// ---------------------------------------------------------------------------
// EpisodicMemory write-op for MI455X (gfx1250, wave32, WMMA).
// Dominant cost: 8 scan steps x 3 Newton-Schulz iters of batched GEMMs
// (64 x [512x512, K=896] and 64 x [896x512, K=512]) ~= 1.4 TFLOP.
// Strategy: all GEMM operands kept/produced as bf16 side-copies so the WMMA
// kernel streams bf16 -> LDS with pure 16B copies (no per-tile conversion
// VALU), double-buffered LDS, 64x32 wave tiles (8 v_wmma per wave per k-step),
// f32 accumulation, f32 (+optional bf16) epilogue.
// ---------------------------------------------------------------------------

#define MEMN 512
#define CODEN 896
#define HIDN 448
#define BSZ 64
#define TT 8
#define EPSF 1e-6f
#define NSTD 0.1f
#define ALPHAF 0.0005f

typedef __attribute__((ext_vector_type(16))) __bf16 v16bf;
typedef __attribute__((ext_vector_type(8)))  float  v8f;

__device__ __forceinline__ unsigned short f2bf(float f) {
    union { float f; unsigned u; } c; c.f = f;
    unsigned u = c.u;
    unsigned r = u + 0x7FFFu + ((u >> 16) & 1u);   // round-to-nearest-even
    return (unsigned short)(r >> 16);
}
__device__ __forceinline__ unsigned pack2(float lo, float hi) {
    return (unsigned)f2bf(lo) | ((unsigned)f2bf(hi) << 16);
}
__device__ __forceinline__ float sigf(float x) { return 1.0f / (1.0f + expf(-x)); }

union PunBF { unsigned u[8]; v16bf v; };

// ---------------------------------------------------------------------------
// Batched GEMM: D = alpha * A @ B + beta * C
//   A: M x K bf16 row-major.  B: K x N bf16 row-major (K-major).
//   C: M x N f32 (read iff beta != 0).  D: f32 out (optional).
//   Dbf: bf16 out (optional) -- lets the next GEMM consume this result
//        without a separate conversion pass.
// Block: 256 threads = 8 waves (2x4). Tile BM=128, BN=128, BK=32.
// Each wave owns 64x32 = 4x2 wmma tiles -> 8 v_wmma_f32_16x16x32_bf16/k-step.
// Double-buffered LDS: fetch k+1 to regs, compute k, store, one barrier.
// ---------------------------------------------------------------------------
__global__ __launch_bounds__(256)
void gemm_bf16_wmma128(const unsigned short* __restrict__ Ag,
                       const unsigned short* __restrict__ Bg,
                       const float* __restrict__ Cg,
                       float* __restrict__ Dg,
                       unsigned short* __restrict__ Dbf,
                       int M, int N, int K,
                       long long sA, long long sB, long long sC, long long sD,
                       float alpha, float beta) {
    constexpr int BM = 128, BN = 128, BK = 32;
    __shared__ unsigned As[2][BM * BK / 2];   // [m][k] pairs: 16 uints per row
    __shared__ unsigned Bs[2][BK * BN / 2];   // [k][n] pairs: 64 uints per row

    const int b = blockIdx.z;
    Ag += (long long)b * sA;
    Bg += (long long)b * sB;
    if (Cg)  Cg  += (long long)b * sC;
    if (Dg)  Dg  += (long long)b * sD;
    if (Dbf) Dbf += (long long)b * sD;

    const int m0 = blockIdx.y * BM;
    const int n0 = blockIdx.x * BN;
    const int tid = threadIdx.x;
    const int lane = tid & 31;
    const int wid = tid >> 5;
    const int wm = (wid >> 2) * 64;      // 2 wave rows
    const int wn = (wid & 3) * 32;       // 4 wave cols
    const int half = lane >> 4;

    // staging-thread mapping
    const int arow = tid >> 2;           // 64 rows/pass (2 passes)
    const int acol = (tid & 3) * 4;      // uint offset within 16-uint row
    const int brow = tid >> 4;           // 16 k/pass (2 passes)
    const int bcol = (tid & 15) * 4;     // uint offset within 64-uint row

    v8f acc[4][2];
#pragma unroll
    for (int i = 0; i < 4; ++i)
#pragma unroll
        for (int j = 0; j < 2; ++j) acc[i][j] = v8f{};

    uint4 ra[2], rb[2];
    const int nk = K / BK;

    // ---- fetch helpers (inlined manually) ----
    auto fetch = [&](int kt, uint4* fa, uint4* fb) {
        const int k0 = kt * BK;
#pragma unroll
        for (int p = 0; p < 2; ++p) {
            const int gm = m0 + p * 64 + arow;
            uint4 v = make_uint4(0u, 0u, 0u, 0u);
            if (gm < M) {
                const unsigned short* ap = Ag + (long long)gm * K + k0 + acol * 2;
                v = *(const uint4*)ap;
                if (kt + 2 < nk) __builtin_prefetch(ap + 2 * BK, 0, 0);
            }
            fa[p] = v;
        }
#pragma unroll
        for (int p = 0; p < 2; ++p) {
            const int gk = k0 + p * 16 + brow;
            const int gn = n0 + bcol * 2;
            uint4 v = make_uint4(0u, 0u, 0u, 0u);
            if (gn < N) v = *(const uint4*)(Bg + (long long)gk * N + gn);
            fb[p] = v;
        }
    };
    auto stage = [&](int buf, const uint4* fa, const uint4* fb) {
#pragma unroll
        for (int p = 0; p < 2; ++p)
            *(uint4*)&As[buf][(p * 64 + arow) * 16 + acol] = fa[p];
#pragma unroll
        for (int p = 0; p < 2; ++p)
            *(uint4*)&Bs[buf][(p * 16 + brow) * 64 + bcol] = fb[p];
    };

    fetch(0, ra, rb);
    stage(0, ra, rb);
    __syncthreads();

    for (int kt = 0; kt < nk; ++kt) {
        const int cur = kt & 1;
        const bool more = (kt + 1) < nk;
        if (more) fetch(kt + 1, ra, rb);

        // fragments per CDNA5 wave32 WMMA layout
        PunBF af[4], bfr[2];
#pragma unroll
        for (int i = 0; i < 4; ++i) {
            const int row = wm + 16 * i + (lane & 15);
#pragma unroll
            for (int v = 0; v < 8; ++v)
                af[i].u[v] = As[cur][row * 16 + 8 * (v >> 2) + (v & 3) + 4 * half];
        }
#pragma unroll
        for (int j = 0; j < 2; ++j) {
            const unsigned short* bsh = (const unsigned short*)Bs[cur];
            const int nc = wn + 16 * j + (lane & 15);
#pragma unroll
            for (int v = 0; v < 8; ++v) {
                const int kk = 16 * (v >> 2) + 8 * half + 2 * (v & 3);
                bfr[j].u[v] = (unsigned)bsh[kk * 128 + nc] | ((unsigned)bsh[(kk + 1) * 128 + nc] << 16);
            }
        }
#pragma unroll
        for (int i = 0; i < 4; ++i)
#pragma unroll
            for (int j = 0; j < 2; ++j)
                acc[i][j] = __builtin_amdgcn_wmma_f32_16x16x32_bf16(
                    false, af[i].v, false, bfr[j].v, (short)0, acc[i][j], false, false);

        if (more) stage(1 - cur, ra, rb);
        __syncthreads();
    }

    // ---- epilogue ----
#pragma unroll
    for (int i = 0; i < 4; ++i) {
#pragma unroll
        for (int j = 0; j < 2; ++j) {
            const int col = n0 + wn + 16 * j + (lane & 15);
            if (col < N) {
#pragma unroll
                for (int r = 0; r < 8; ++r) {
                    const int row = m0 + wm + 16 * i + r + 8 * half;
                    if (row < M) {
                        const long long off = (long long)row * N + col;
                        float v = alpha * acc[i][j][r];
                        if (beta != 0.0f) v += beta * Cg[off];
                        if (Dg)  Dg[off]  = v;
                        if (Dbf) Dbf[off] = f2bf(v);
                    }
                }
            }
        }
    }
}

// ---------------------------------------------------------------------------
// f32 -> bf16 bulk convert (n multiple of 4).
// ---------------------------------------------------------------------------
__global__ void cvt_bf16_kernel(const float* __restrict__ s,
                                unsigned short* __restrict__ d, long long n) {
    const long long i = ((long long)blockIdx.x * blockDim.x + threadIdx.x) * 4;
    if (i >= n) return;
    const float4 v = *(const float4*)(s + i);
    *(uint2*)(d + i) = make_uint2(pack2(v.x, v.y), pack2(v.z, v.w));
}

// Transpose + convert: src (R x C) f32 -> dst (C x R) bf16.
__global__ __launch_bounds__(256)
void transpose_cvt_kernel(const float* __restrict__ src,
                          unsigned short* __restrict__ dst, int R, int C) {
    __shared__ float tile[32][33];
    const int r0 = blockIdx.y * 32, c0 = blockIdx.x * 32;
    const int tx = threadIdx.x & 31, ty = threadIdx.x >> 5;
#pragma unroll
    for (int p = 0; p < 4; ++p) {
        const int r = r0 + ty + p * 8, c = c0 + tx;
        tile[ty + p * 8][tx] = (r < R && c < C) ? src[(long long)r * C + c] : 0.f;
    }
    __syncthreads();
#pragma unroll
    for (int p = 0; p < 4; ++p) {
        const int c = c0 + ty + p * 8, r = r0 + tx;
        if (c < C && r < R) dst[(long long)c * R + r] = f2bf(tile[tx][ty + p * 8]);
    }
}

// P0 = ALPHA * mean^T per batch: f32 and bf16 outputs together.
__global__ __launch_bounds__(256)
void transpose_scale_bf_kernel(const float* __restrict__ src,
                               float* __restrict__ dstf,
                               unsigned short* __restrict__ dstbf,
                               int R, int C, float scale) {
    __shared__ float tile[32][33];
    const int b = blockIdx.z;
    src   += (long long)b * R * C;
    dstf  += (long long)b * R * C;
    dstbf += (long long)b * R * C;
    const int r0 = blockIdx.y * 32, c0 = blockIdx.x * 32;
    const int tx = threadIdx.x & 31, ty = threadIdx.x >> 5;
#pragma unroll
    for (int p = 0; p < 4; ++p) {
        const int r = r0 + ty + p * 8, c = c0 + tx;
        tile[ty + p * 8][tx] = (r < R && c < C) ? src[(long long)r * C + c] : 0.f;
    }
    __syncthreads();
#pragma unroll
    for (int p = 0; p < 4; ++p) {
        const int c = c0 + ty + p * 8, r = r0 + tx;
        if (c < C && r < R) {
            const float v = scale * tile[tx][ty + p * 8];
            dstf[(long long)c * R + r]  = v;
            dstbf[(long long)c * R + r] = f2bf(v);
        }
    }
}

// Deterministic hash-based standard normal (stands in for jax PRNG noise).
__device__ __forceinline__ unsigned pcg(unsigned v) {
    v = v * 747796405u + 2891336453u;
    unsigned w = ((v >> ((v >> 28) + 4u)) ^ v) * 277803737u;
    return (w >> 22) ^ w;
}
__global__ void noise_kernel(float* __restrict__ out, int n) {
    const int i = blockIdx.x * blockDim.x + threadIdx.x;
    if (i >= n) return;
    const unsigned a = pcg((unsigned)i * 2u + 1u), b = pcg((unsigned)i * 2u + 2u);
    const float u1 = ((float)a + 1.0f) * 2.3283064365386963e-10f;
    const float u2 = (float)b * 2.3283064365386963e-10f;
    out[i] = sqrtf(-2.f * logf(u1)) * cosf(6.28318530717958647f * u2);
}

__global__ void init_mean_kernel(float* __restrict__ mean, const float* __restrict__ mm) {
    const long long i = (long long)blockIdx.x * blockDim.x + threadIdx.x;
    if (i >= (long long)BSZ * MEMN * CODEN) return;
    mean[i] = mm[i % ((long long)MEMN * CODEN)];
}
__global__ void init_cov_kernel(float* __restrict__ cov) {
    const long long i = (long long)blockIdx.x * blockDim.x + threadIdx.x;
    if (i >= (long long)BSZ * MEMN * MEMN) return;
    const int n = (int)(i % MEMN);
    const int m = (int)((i / MEMN) % MEMN);
    cov[i] = (m == n) ? (1.0f + EPSF) : 0.0f;
}

// w[b][m] = sum_d (zp[t][b][d] + 0.1*noise[t][b][d]) * Pinv[b][d][m]
__global__ __launch_bounds__(512)
void wvec_kernel(const float* __restrict__ zp, const float* __restrict__ nz,
                 const float* __restrict__ Pinv, float* __restrict__ w, int t) {
    __shared__ float zs[CODEN];
    const int b = blockIdx.x, m = threadIdx.x;
    const float* z = zp + ((long long)t * BSZ + b) * CODEN;
    const float* nn = nz + ((long long)t * BSZ + b) * CODEN;
    for (int d = m; d < CODEN; d += 512) zs[d] = z[d] + NSTD * nn[d];
    __syncthreads();
    const float* P = Pinv + (long long)b * CODEN * MEMN;
    float acc = 0.f;
    for (int d = 0; d < CODEN; ++d) acc = fmaf(zs[d], P[(long long)d * MEMN + m], acc);
    w[b * MEMN + m] = acc;
}

// delta[b][d] = zp[t][b][d] - sum_m w[b][m]*mean[b][m][d]
__global__ __launch_bounds__(896)
void delta_kernel(const float* __restrict__ zp, const float* __restrict__ w,
                  const float* __restrict__ mean, float* __restrict__ delta, int t) {
    __shared__ float ws_[MEMN];
    const int b = blockIdx.x, d = threadIdx.x;
    if (d < MEMN) ws_[d] = w[b * MEMN + d];
    __syncthreads();
    const float* Mb = mean + (long long)b * MEMN * CODEN;
    float acc = 0.f;
    for (int m = 0; m < MEMN; ++m) acc = fmaf(ws_[m], Mb[(long long)m * CODEN + d], acc);
    delta[b * CODEN + d] = zp[((long long)t * BSZ + b) * CODEN + d] - acc;
}

// wU[b][n] = sum_m w[b][m]*cov[b][m][n]
__global__ __launch_bounds__(512)
void wu_kernel(const float* __restrict__ w, const float* __restrict__ cov,
               float* __restrict__ wU) {
    __shared__ float ws_[MEMN];
    const int b = blockIdx.x, n = threadIdx.x;
    ws_[n] = w[b * MEMN + n];
    __syncthreads();
    const float* Cb = cov + (long long)b * MEMN * MEMN;
    float acc = 0.f;
    for (int m = 0; m < MEMN; ++m) acc = fmaf(ws_[m], Cb[(long long)m * MEMN + n], acc);
    wU[b * MEMN + n] = acc;
}

__global__ __launch_bounds__(256)
void sigma_kernel(const float* __restrict__ w, const float* __restrict__ wU,
                  float* __restrict__ sigma) {
    __shared__ float red[256];
    const int b = blockIdx.x, t = threadIdx.x;
    float acc = 0.f;
    for (int m = t; m < MEMN; m += 256) acc += w[b * MEMN + m] * wU[b * MEMN + m];
    red[t] = acc; __syncthreads();
    for (int s = 128; s > 0; s >>= 1) { if (t < s) red[t] += red[t + s]; __syncthreads(); }
    if (t == 0) sigma[b] = red[0] + NSTD * NSTD;
}

__global__ void mean_update_kernel(float* __restrict__ mean, const float* __restrict__ wU,
                                   const float* __restrict__ sigma, const float* __restrict__ delta) {
    const long long i = (long long)blockIdx.x * blockDim.x + threadIdx.x;
    if (i >= (long long)BSZ * MEMN * CODEN) return;
    const int d = (int)(i % CODEN);
    const long long t2 = i / CODEN;
    const int m = (int)(t2 % MEMN);
    const int b = (int)(t2 / MEMN);
    const float c = wU[b * MEMN + m] / sigma[b];
    mean[i] += c * delta[b * CODEN + d];
}
__global__ void cov_update_kernel(float* __restrict__ cov, const float* __restrict__ wU,
                                  const float* __restrict__ sigma) {
    const long long i = (long long)blockIdx.x * blockDim.x + threadIdx.x;
    if (i >= (long long)BSZ * MEMN * MEMN) return;
    const int n = (int)(i % MEMN);
    const long long t2 = i / MEMN;
    const int m = (int)(t2 % MEMN);
    const int b = (int)(t2 / MEMN);
    const float c = wU[b * MEMN + m] / sigma[b];
    cov[i] -= c * wU[b * MEMN + n];
}

// LSTM cell pointwise update; writes h (f32 + bf16) and concat output.
__global__ __launch_bounds__(448)
void lstm_cell_kernel(const float* __restrict__ G, const float* __restrict__ bih,
                      const float* __restrict__ bhh, float* __restrict__ h,
                      unsigned short* __restrict__ hbf, float* __restrict__ c,
                      float* __restrict__ xout, unsigned short* __restrict__ xoutbf,
                      int t, int dirOffset) {
    const int b = blockIdx.x, j = threadIdx.x;
    const float* g = G + b * 4 * HIDN;
    const float i_ = g[j]             + bih[j]             + bhh[j];
    const float f_ = g[HIDN + j]      + bih[HIDN + j]      + bhh[HIDN + j];
    const float gg = g[2 * HIDN + j]  + bih[2 * HIDN + j]  + bhh[2 * HIDN + j];
    const float o_ = g[3 * HIDN + j]  + bih[3 * HIDN + j]  + bhh[3 * HIDN + j];
    const float cn = sigf(f_) * c[b * HIDN + j] + sigf(i_) * tanhf(gg);
    const float hn = sigf(o_) * tanhf(cn);
    c[b * HIDN + j] = cn;
    h[b * HIDN + j] = hn;
    hbf[b * HIDN + j] = f2bf(hn);
    const long long off = ((long long)t * BSZ + b) * (2 * HIDN) + dirOffset + j;
    xout[off] = hn;
    if (xoutbf) xoutbf[off] = f2bf(hn);
}

__global__ __launch_bounds__(256)
void dkl_partial_kernel(const float* __restrict__ mean, const float* __restrict__ cov,
                        const float* __restrict__ mm, float* __restrict__ part) {
    const int b = blockIdx.x, t = threadIdx.x;
    const float pd = 1.0f + EPSF;
    float s1 = 0.f, s4 = 0.f;
    for (int m = t; m < MEMN; m += 256) {
        const float qd = cov[(long long)b * MEMN * MEMN + (long long)m * (MEMN + 1)];
        s1 += qd / pd;
        s4 += logf(pd) - logf(qd);
    }
    float s2 = 0.f;
    const float* Mb = mean + (long long)b * MEMN * CODEN;
    for (long long i = t; i < (long long)MEMN * CODEN; i += 256) {
        const float df = Mb[i] - mm[i];
        s2 += df * df / pd;
    }
    __shared__ float r1[256], r2[256], r4[256];
    r1[t] = s1; r2[t] = s2; r4[t] = s4; __syncthreads();
    for (int s = 128; s > 0; s >>= 1) {
        if (t < s) { r1[t] += r1[t + s]; r2[t] += r2[t + s]; r4[t] += r4[t + s]; }
        __syncthreads();
    }
    if (t == 0)
        part[b] = (float)CODEN * r1[0] + r2[0] - (float)CODEN * (float)MEMN + (float)CODEN * r4[0];
}
__global__ void dkl_final_kernel(const float* __restrict__ part, float* __restrict__ out) {
    if (threadIdx.x == 0 && blockIdx.x == 0) {
        float s = 0.f;
        for (int b = 0; b < BSZ; ++b) s += part[b];
        out[0] = s / (float)BSZ;
    }
}

// ---------------------------------------------------------------------------
// Host orchestration
// ---------------------------------------------------------------------------
static inline void launch_gemm(hipStream_t s,
                               const unsigned short* A, const unsigned short* B,
                               const float* C, float* D, unsigned short* Dbf,
                               int M, int N, int K,
                               long long sA, long long sB, long long sC, long long sD,
                               float alpha, float beta, int batch) {
    dim3 grid((N + 127) / 128, (M + 127) / 128, batch);
    gemm_bf16_wmma128<<<grid, 256, 0, s>>>(A, B, C, D, Dbf, M, N, K, sA, sB, sC, sD, alpha, beta);
}

extern "C" void kernel_launch(void* const* d_in, const int* in_sizes, int n_in,
                              void* d_out, int out_size, void* d_ws, size_t ws_size,
                              hipStream_t stream) {
    (void)in_sizes; (void)n_in; (void)out_size; (void)ws_size;
    const float* Z  = (const float*)d_in[0];
    const float* MM = (const float*)d_in[1];
    const float* WIH[4] = {(const float*)d_in[2], (const float*)d_in[6], (const float*)d_in[10], (const float*)d_in[14]};
    const float* WHH[4] = {(const float*)d_in[3], (const float*)d_in[7], (const float*)d_in[11], (const float*)d_in[15]};
    const float* BIH[4] = {(const float*)d_in[4], (const float*)d_in[8], (const float*)d_in[12], (const float*)d_in[16]};
    const float* BHH[4] = {(const float*)d_in[5], (const float*)d_in[9], (const float*)d_in[13], (const float*)d_in[17]};

    float* mean = (float*)d_out;                                   // (64,512,896)
    float* cov  = mean + (long long)BSZ * MEMN * CODEN;            // (64,512,512)
    float* dkl  = cov + (long long)BSZ * MEMN * MEMN;              // scalar

    // ---- f32 workspace ----
    float* ws = (float*)d_ws;
    float* P0 = ws;  ws += (long long)BSZ * CODEN * MEMN;
    float* P1 = ws;  ws += (long long)BSZ * CODEN * MEMN;
    float* XW = ws;  ws += (long long)TT * BSZ * 4 * HIDN;
    float* X1 = ws;  ws += (long long)TT * BSZ * CODEN;
    float* ZP = ws;  ws += (long long)TT * BSZ * CODEN;
    float* NZ = ws;  ws += (long long)TT * BSZ * CODEN;
    float* G  = ws;  ws += (long long)BSZ * 4 * HIDN;
    float* Hb = ws;  ws += (long long)BSZ * HIDN;
    float* Cb = ws;  ws += (long long)BSZ * HIDN;
    float* Wv = ws;  ws += (long long)BSZ * MEMN;
    float* WU = ws;  ws += (long long)BSZ * MEMN;
    float* DE = ws;  ws += (long long)BSZ * CODEN;
    float* SI = ws;  ws += 64;
    float* PA = ws;  ws += 64;
    // ---- bf16 workspace (2-byte elems, carved on 16B boundaries) ----
    unsigned short* bw = (unsigned short*)ws;
    unsigned short* MEANbf = bw;  bw += (long long)BSZ * MEMN * CODEN;
    unsigned short* P0bf   = bw;  bw += (long long)BSZ * CODEN * MEMN;
    unsigned short* P1bf   = bw;  bw += (long long)BSZ * CODEN * MEMN;
    unsigned short* APbf   = bw;  bw += (long long)BSZ * MEMN * MEMN;
    unsigned short* Zbf    = bw;  bw += (long long)TT * BSZ * CODEN;
    unsigned short* X1bf   = bw;  bw += (long long)TT * BSZ * CODEN;
    unsigned short* Hbf    = bw;  bw += (long long)BSZ * HIDN;
    unsigned short* WIHT[4], *WHHT[4];
    for (int i = 0; i < 4; ++i) { WIHT[i] = bw; bw += (long long)CODEN * 4 * HIDN; }
    for (int i = 0; i < 4; ++i) { WHHT[i] = bw; bw += (long long)HIDN * 4 * HIDN; }

    // ---- one-time prep: noise, output init, bf16 operand prep ----
    {
        const int n = TT * BSZ * CODEN;
        noise_kernel<<<(n + 255) / 256, 256, 0, stream>>>(NZ, n);
        const long long nm = (long long)BSZ * MEMN * CODEN;
        init_mean_kernel<<<(unsigned)((nm + 255) / 256), 256, 0, stream>>>(mean, MM);
        const long long nc = (long long)BSZ * MEMN * MEMN;
        init_cov_kernel<<<(unsigned)((nc + 255) / 256), 256, 0, stream>>>(cov);
        cvt_bf16_kernel<<<(unsigned)((n / 4 + 255) / 256), 256, 0, stream>>>(Z, Zbf, n);
        for (int i = 0; i < 4; ++i) {
            dim3 g1(CODEN / 32, (4 * HIDN) / 32);
            transpose_cvt_kernel<<<g1, 256, 0, stream>>>(WIH[i], WIHT[i], 4 * HIDN, CODEN);
            dim3 g2(HIDN / 32, (4 * HIDN) / 32);
            transpose_cvt_kernel<<<g2, 256, 0, stream>>>(WHH[i], WHHT[i], 4 * HIDN, HIDN);
        }
    }

    // ---- BiLSTM (2 layers x 2 directions) ----
    for (int ld = 0; ld < 4; ++ld) {
        const int layer = ld >> 1, rev = ld & 1;
        const unsigned short* xin = (layer == 0) ? Zbf : X1bf;
        float* xout = (layer == 0) ? X1 : ZP;
        unsigned short* xoutbf = (layer == 0) ? X1bf : nullptr;  // ZP consumed in f32 only
        // XW = x @ wih^T for all timesteps: (512 x 1792), K=896
        launch_gemm(stream, xin, WIHT[ld], nullptr, XW, nullptr,
                    TT * BSZ, 4 * HIDN, CODEN, 0, 0, 0, 0, 1.0f, 0.0f, 1);
        hipMemsetAsync(Hb, 0, sizeof(float) * BSZ * HIDN, stream);
        hipMemsetAsync(Cb, 0, sizeof(float) * BSZ * HIDN, stream);
        hipMemsetAsync(Hbf, 0, sizeof(unsigned short) * BSZ * HIDN, stream);
        for (int s = 0; s < TT; ++s) {
            const int t = rev ? (TT - 1 - s) : s;
            // G = h @ whh^T + XW[t]   (64 x 1792), K=448
            launch_gemm(stream, Hbf, WHHT[ld], XW + (long long)t * BSZ * 4 * HIDN, G, nullptr,
                        BSZ, 4 * HIDN, HIDN, 0, 0, 0, 0, 1.0f, 1.0f, 1);
            lstm_cell_kernel<<<BSZ, HIDN, 0, stream>>>(G, BIH[ld], BHH[ld], Hb, Hbf, Cb,
                                                       xout, xoutbf, t, rev ? HIDN : 0);
        }
    }

    // ---- scan over 8 timesteps ----
    const long long nm = (long long)BSZ * MEMN * CODEN;
    for (int t = 0; t < TT; ++t) {
        cvt_bf16_kernel<<<(unsigned)((nm / 4 + 255) / 256), 256, 0, stream>>>(mean, MEANbf, nm);
        {   // P = ALPHA * mean^T (f32 + bf16)
            dim3 grid(CODEN / 32, MEMN / 32, BSZ);
            transpose_scale_bf_kernel<<<grid, 256, 0, stream>>>(mean, P0, P0bf, MEMN, CODEN, ALPHAF);
        }
        float* Pc = P0; float* Po = P1;
        unsigned short* Pcbf = P0bf; unsigned short* Pobf = P1bf;
        for (int it = 0; it < 3; ++it) {
            // AP(bf16) = mean @ P : 64 x [512 x 512, K=896]; f32 result unneeded
            launch_gemm(stream, MEANbf, Pcbf, nullptr, nullptr, APbf,
                        MEMN, MEMN, CODEN,
                        (long long)MEMN * CODEN, (long long)CODEN * MEMN, 0, (long long)MEMN * MEMN,
                        1.0f, 0.0f, BSZ);
            // P' = 2P - P @ AP : 64 x [896 x 512, K=512]; f32 + bf16 outputs
            launch_gemm(stream, Pcbf, APbf, Pc, Po, Pobf,
                        CODEN, MEMN, MEMN,
                        (long long)CODEN * MEMN, (long long)MEMN * MEMN,
                        (long long)CODEN * MEMN, (long long)CODEN * MEMN,
                        -1.0f, 2.0f, BSZ);
            float* tf = Pc; Pc = Po; Po = tf;
            unsigned short* tb = Pcbf; Pcbf = Pobf; Pobf = tb;
        }
        wvec_kernel<<<BSZ, 512, 0, stream>>>(ZP, NZ, Pc, Wv, t);
        delta_kernel<<<BSZ, 896, 0, stream>>>(ZP, Wv, mean, DE, t);
        wu_kernel<<<BSZ, 512, 0, stream>>>(Wv, cov, WU);
        sigma_kernel<<<BSZ, 256, 0, stream>>>(Wv, WU, SI);
        mean_update_kernel<<<(unsigned)((nm + 255) / 256), 256, 0, stream>>>(mean, WU, SI, DE);
        const long long nc = (long long)BSZ * MEMN * MEMN;
        cov_update_kernel<<<(unsigned)((nc + 255) / 256), 256, 0, stream>>>(cov, WU, SI);
    }

    // ---- KL divergence ----
    dkl_partial_kernel<<<BSZ, 256, 0, stream>>>(mean, cov, MM, PA);
    dkl_final_kernel<<<1, 32, 0, stream>>>(PA, dkl);
}
